// NSAAttention_11355893530935
// MI455X (gfx1250) — matrix operand order repaired and checked by reading, not compiled, not run on hardware
//
#include <hip/hip_runtime.h>
#include <hip/hip_bf16.h>
#include <math.h>

#define DEV __device__ __forceinline__

typedef __attribute__((ext_vector_type(16))) _Float16 v16h;
typedef __attribute__((ext_vector_type(8)))  float    v8f;

namespace {

constexpr int S_    = 1536;
constexpr int DIMc  = 1024;
constexpr int Gc    = 2;
constexpr int Hc    = 16;
constexpr int HPGc  = 8;
constexpr int Lc    = 32;   // compress window
constexpr int Dstr  = 16;   // compress stride
constexpr int NSELc = 16;
constexpr int Wc    = 512;
constexpr int NCc   = 95;   // (S-L)/D + 1
constexpr int NCPc  = 96;   // padded
constexpr int NBc   = 24;   // S / LSEL
constexpr int HIDc  = 32;
constexpr float SCALEc = 0.125f;  // 1/sqrt(64)
constexpr float NEGc   = -1e30f;

union HFrag { v16h h; unsigned u[8]; };

DEV int a_kb(int v, int hi) { int k = (v < 4) ? (2 * v) : (16 + 2 * (v - 4)); return k + (hi ? 8 : 0); }

// A fragment (16x32 f16): lane m = lane&15 (both halves), K split per ISA table.
DEV v16h load_a16(const _Float16* p, int ld, int lane) {
  const int hi = lane >> 4, m = lane & 15;
  HFrag f;
  const unsigned* row = (const unsigned*)(p + (size_t)m * ld);
#pragma unroll
  for (int v = 0; v < 8; ++v) f.u[v] = row[a_kb(v, hi) >> 1];
  return f.h;
}

// B fragment (32x16 f16) from Bt[n][k] row-major: K split per ISA table (same as A).
DEV v16h load_b16(const _Float16* p, int ld, int lane) {
  const int hi = lane >> 4, n = lane & 15;
  HFrag f;
  const unsigned* row = (const unsigned*)(p + (size_t)n * ld);
#pragma unroll
  for (int v = 0; v < 8; ++v) f.u[v] = row[a_kb(v, hi) >> 1];
  return f.h;
}

DEV v8f wmma_f16(v16h a, v16h b, v8f c) {
  return __builtin_amdgcn_wmma_f32_16x16x32_f16(false, a, false, b, (short)0, c, false, false);
}

DEV v8f vzero() {
  v8f z;
#pragma unroll
  for (int i = 0; i < 8; ++i) z[i] = 0.0f;
  return z;
}

DEV float redmax16(float x) {
#pragma unroll
  for (int o = 1; o < 16; o <<= 1) x = fmaxf(x, __shfl_xor(x, o, 32));
  return x;
}
DEV float redsum16(float x) {
#pragma unroll
  for (int o = 1; o < 16; o <<= 1) x += __shfl_xor(x, o, 32);
  return x;
}
DEV unsigned redor32(unsigned x) {
#pragma unroll
  for (int o = 1; o < 32; o <<= 1) x |= (unsigned)__shfl_xor((int)x, o, 32);
  return x;
}

} // namespace

// ---------------- f32 -> f16 convert ----------------
__global__ void k_cvt_f16(const float* __restrict__ s, _Float16* __restrict__ d, int n) {
  int i = blockIdx.x * blockDim.x + threadIdx.x;
  if (i < n) d[i] = (_Float16)s[i];
}

// ---------------- generic WMMA GEMM: C[M,N] = A[M,K] * B[N,K]^T ----------------
__global__ void __launch_bounds__(32) k_gemm(const _Float16* __restrict__ A,
                                             const _Float16* __restrict__ B,
                                             float* __restrict__ C, int M, int N, int K) {
  const int lane = threadIdx.x;
  const int nt = blockIdx.x * 64;
  const int mt = blockIdx.y * 16;
  v8f acc0 = vzero(), acc1 = vzero(), acc2 = vzero(), acc3 = vzero();
  const _Float16* arow = A + (size_t)mt * K;
  for (int k0 = 0; k0 < K; k0 += 32) {
    v16h a  = load_a16(arow + k0, K, lane);
    v16h b0 = load_b16(B + (size_t)(nt +  0) * K + k0, K, lane);
    v16h b1 = load_b16(B + (size_t)(nt + 16) * K + k0, K, lane);
    v16h b2 = load_b16(B + (size_t)(nt + 32) * K + k0, K, lane);
    v16h b3 = load_b16(B + (size_t)(nt + 48) * K + k0, K, lane);
    acc0 = wmma_f16(a, b0, acc0);
    acc1 = wmma_f16(a, b1, acc1);
    acc2 = wmma_f16(a, b2, acc2);
    acc3 = wmma_f16(a, b3, acc3);
  }
  const int hi = lane >> 4, n = lane & 15;
#pragma unroll
  for (int r = 0; r < 8; ++r) {
    int m = mt + r + (hi ? 8 : 0);
    float* crow = C + (size_t)m * N + nt + n;
    crow[0] = acc0[r]; crow[16] = acc1[r]; crow[32] = acc2[r]; crow[48] = acc3[r];
  }
}

// ---------------- RoPE on Q + group-mean (qp) ----------------
__global__ void k_rope_q(const float* __restrict__ qf, _Float16* __restrict__ qh,
                         float* __restrict__ qp) {
  int i = blockIdx.x * blockDim.x + threadIdx.x;  // S*32
  if (i >= S_ * 32) return;
  int s = i >> 5, d2 = i & 31;
  float inv = __powf(10000.0f, -(float)d2 / 32.0f);
  float ang = (float)s * inv;
  float cs = __cosf(ang), sn = __sinf(ang);
  float accA[2] = {0.f, 0.f}, accB[2] = {0.f, 0.f};
#pragma unroll
  for (int h = 0; h < Hc; ++h) {
    float x1 = qf[(size_t)s * 1024 + h * 64 + d2];
    float x2 = qf[(size_t)s * 1024 + h * 64 + d2 + 32];
    float o1 = x1 * cs - x2 * sn;
    float o2 = x2 * cs + x1 * sn;
    qh[((size_t)s * Hc + h) * 64 + d2]      = (_Float16)o1;
    qh[((size_t)s * Hc + h) * 64 + d2 + 32] = (_Float16)o2;
    accA[h >> 3] += o1; accB[h >> 3] += o2;
  }
#pragma unroll
  for (int g = 0; g < Gc; ++g) {
    qp[((size_t)s * Gc + g) * 64 + d2]      = accA[g] * 0.125f;
    qp[((size_t)s * Gc + g) * 64 + d2 + 32] = accB[g] * 0.125f;
  }
}

// ---------------- RoPE K (sel/win/cmp) + pack V (transpose sel/win) ----------------
__global__ void k_rope_kv(const float* __restrict__ kvf,
                          _Float16* __restrict__ ksel, _Float16* __restrict__ kwin,
                          _Float16* __restrict__ kcmp, _Float16* __restrict__ vselt,
                          _Float16* __restrict__ vwint, _Float16* __restrict__ vcmpr) {
  int i = blockIdx.x * blockDim.x + threadIdx.x;  // S*G*32
  if (i >= S_ * Gc * 32) return;
  int d2 = i & 31, g = (i >> 5) & 1, s = i >> 6;
  float inv = __powf(10000.0f, -(float)d2 / 32.0f);
  float ang = (float)s * inv;
  float cs = __cosf(ang), sn = __sinf(ang);
  const float* base = kvf + (size_t)s * 768;
  // stacked rows: Ksel 0 | Vsel 128 | Kwin 256 | Vwin 384 | Kcmp 512 | Vcmp 640
  const int ko[3] = {0, 256, 512};
  _Float16* kout[3] = {ksel, kwin, kcmp};
#pragma unroll
  for (int j = 0; j < 3; ++j) {
    float x1 = base[ko[j] + g * 64 + d2];
    float x2 = base[ko[j] + g * 64 + d2 + 32];
    kout[j][((size_t)g * S_ + s) * 64 + d2]      = (_Float16)(x1 * cs - x2 * sn);
    kout[j][((size_t)g * S_ + s) * 64 + d2 + 32] = (_Float16)(x2 * cs + x1 * sn);
  }
  vselt[((size_t)g * 64 + d2) * S_ + s]      = (_Float16)base[128 + g * 64 + d2];
  vselt[((size_t)g * 64 + d2 + 32) * S_ + s] = (_Float16)base[128 + g * 64 + d2 + 32];
  vwint[((size_t)g * 64 + d2) * S_ + s]      = (_Float16)base[384 + g * 64 + d2];
  vwint[((size_t)g * 64 + d2 + 32) * S_ + s] = (_Float16)base[384 + g * 64 + d2 + 32];
  vcmpr[((size_t)g * S_ + s) * 64 + d2]      = (_Float16)base[640 + g * 64 + d2];
  vcmpr[((size_t)g * S_ + s) * 64 + d2 + 32] = (_Float16)base[640 + g * 64 + d2 + 32];
}

// ---------------- compress pooling (mean over L=32, stride 16), pad to 96 ----------------
__global__ void k_cmp_pool(const _Float16* __restrict__ kcmpr, const _Float16* __restrict__ vcmpr,
                           _Float16* __restrict__ kpool, _Float16* __restrict__ vt) {
  int i = blockIdx.x * blockDim.x + threadIdx.x;  // G*NCP*64
  if (i >= Gc * NCPc * 64) return;
  int d = i & 63, c = (i >> 6) % NCPc, g = i / (64 * NCPc);
  float ka = 0.f, va = 0.f;
  if (c < NCc) {
    for (int j = 0; j < Lc; ++j) {
      int t = c * Dstr + j;
      ka += (float)kcmpr[((size_t)g * S_ + t) * 64 + d];
      va += (float)vcmpr[((size_t)g * S_ + t) * 64 + d];
    }
    ka *= (1.0f / Lc); va *= (1.0f / Lc);
  }
  kpool[((size_t)g * NCPc + c) * 64 + d] = (_Float16)ka;
  vt[((size_t)g * 64 + d) * NCPc + c]    = (_Float16)va;
}

// ---------------- compressed attention (single pass, 95 keys) ----------------
__global__ void __launch_bounds__(32) k_attn_cmp(
    const _Float16* __restrict__ qh, const _Float16* __restrict__ kpool,
    const _Float16* __restrict__ vt, float* __restrict__ pcmp, float* __restrict__ ocmp) {
  __shared__ _Float16 Pl[16][32];
  const int lane = threadIdx.x;
  const int s0 = blockIdx.x * 16, g = blockIdx.y, hh = blockIdx.z;
  const int h = g * HPGc + hh;
  const int hi = lane >> 4, ln = lane & 15;

  v16h aq0 = load_a16(qh + ((size_t)s0 * Hc + h) * 64, Hc * 64, lane);
  v16h aq1 = load_a16(qh + ((size_t)s0 * Hc + h) * 64 + 32, Hc * 64, lane);

  v8f sc[6];
#pragma unroll
  for (int ti = 0; ti < 6; ++ti) {
    v16h b0 = load_b16(kpool + ((size_t)g * NCPc + ti * 16) * 64, 64, lane);
    v16h b1 = load_b16(kpool + ((size_t)g * NCPc + ti * 16) * 64 + 32, 64, lane);
    v8f a = vzero();
    a = wmma_f16(aq0, b0, a);
    a = wmma_f16(aq1, b1, a);
    int c = ti * 16 + ln;
#pragma unroll
    for (int r = 0; r < 8; ++r) {
      int s = s0 + r + (hi ? 8 : 0);
      bool ok = (c < NCc) && (s >= c * Dstr + Lc - 1);
      a[r] = ok ? a[r] * SCALEc : NEGc;
    }
    sc[ti] = a;
  }
  float Mx[8], invs[8];
#pragma unroll
  for (int r = 0; r < 8; ++r) {
    float m = NEGc;
#pragma unroll
    for (int ti = 0; ti < 6; ++ti) m = fmaxf(m, sc[ti][r]);
    Mx[r] = redmax16(m);
  }
#pragma unroll
  for (int ti = 0; ti < 6; ++ti) {
#pragma unroll
    for (int r = 0; r < 8; ++r) {
      float v = sc[ti][r];
      sc[ti][r] = (v > -1e29f) ? __expf(v - Mx[r]) : 0.0f;
    }
  }
#pragma unroll
  for (int r = 0; r < 8; ++r) {
    float su = 0.f;
#pragma unroll
    for (int ti = 0; ti < 6; ++ti) su += sc[ti][r];
    su = redsum16(su);
    invs[r] = (su > 0.f) ? 1.0f / su : 0.0f;
  }
#pragma unroll
  for (int ti = 0; ti < 6; ++ti) {
    int c = ti * 16 + ln;
#pragma unroll
    for (int r = 0; r < 8; ++r) {
      int s = s0 + r + (hi ? 8 : 0);
      float p = sc[ti][r] * invs[r];
      sc[ti][r] = p;
      pcmp[(((size_t)s * Gc + g) * HPGc + hh) * NCPc + c] = p;
    }
  }
  v8f o0 = vzero(), o1 = vzero(), o2 = vzero(), o3 = vzero();
#pragma unroll
  for (int ch = 0; ch < 3; ++ch) {
    __syncthreads();
#pragma unroll
    for (int half = 0; half < 2; ++half) {
      v8f p = sc[ch * 2 + half];
#pragma unroll
      for (int r = 0; r < 8; ++r) Pl[r + (hi ? 8 : 0)][half * 16 + ln] = (_Float16)p[r];
    }
    __syncthreads();
    v16h ap = load_a16(&Pl[0][0], 32, lane);
    int cb = ch * 32;
    v16h vb0 = load_b16(vt + ((size_t)g * 64 +  0) * NCPc + cb, NCPc, lane);
    v16h vb1 = load_b16(vt + ((size_t)g * 64 + 16) * NCPc + cb, NCPc, lane);
    v16h vb2 = load_b16(vt + ((size_t)g * 64 + 32) * NCPc + cb, NCPc, lane);
    v16h vb3 = load_b16(vt + ((size_t)g * 64 + 48) * NCPc + cb, NCPc, lane);
    o0 = wmma_f16(ap, vb0, o0);
    o1 = wmma_f16(ap, vb1, o1);
    o2 = wmma_f16(ap, vb2, o2);
    o3 = wmma_f16(ap, vb3, o3);
  }
#pragma unroll
  for (int r = 0; r < 8; ++r) {
    int s = s0 + r + (hi ? 8 : 0);
    float* orow = ocmp + ((size_t)s * Hc + h) * 64 + ln;
    orow[0] = o0[r]; orow[16] = o1[r]; orow[32] = o2[r]; orow[48] = o3[r];
  }
}

// ---------------- block scores + top-16 selection -> 24-bit mask ----------------
__global__ void k_pgrp_topk(const float* __restrict__ pcmp, unsigned* __restrict__ selmask) {
  int i = blockIdx.x * blockDim.x + threadIdx.x;  // S*G
  if (i >= S_ * Gc) return;
  int g = i & 1, s = i >> 1;
  float pg[NBc];
#pragma unroll
  for (int m = 0; m < NBc; ++m) pg[m] = 0.f;
  for (int c = 0; c < NCc; ++c) {
    float ps = 0.f;
    const float* base = pcmp + (((size_t)s * Gc + g) * HPGc) * NCPc + c;
#pragma unroll
    for (int hh = 0; hh < HPGc; ++hh) ps += base[(size_t)hh * NCPc];
    int lo = (c * Dstr) >> 6;
    int hb = (c * Dstr + Lc - 1) >> 6;
    for (int m = lo; m <= hb && m < NBc; ++m) {
      int ov = min(c * Dstr + Lc, m * 64 + 64) - max(c * Dstr, m * 64);
      if (ov < 0) ov = 0;
      pg[m] += ps * ((float)ov / (float)Lc);
    }
  }
  int cur = s >> 6;
  float score[NBc];
#pragma unroll
  for (int m = 0; m < NBc; ++m) {
    float sc = pg[m];
    if (m == 0 || m == cur) sc += 1.0e6f;
    if (m * 64 > s) sc = -1.0e9f;
    score[m] = sc;
  }
  unsigned mask = 0;
  for (int k = 0; k < NSELc; ++k) {
    int best = 0; float bv = -3.0e38f;
#pragma unroll
    for (int m = 0; m < NBc; ++m) {
      if (!((mask >> m) & 1) && score[m] > bv) { bv = score[m]; best = m; }
    }
    mask |= (1u << best);
  }
  selmask[i] = mask;
}

// ---------------- selected attention (flash over selected 64-token blocks) ----------------
__global__ void __launch_bounds__(32) k_attn_sel(
    const _Float16* __restrict__ qh, const _Float16* __restrict__ ks,
    const _Float16* __restrict__ vtS, const unsigned* __restrict__ selmask,
    float* __restrict__ osel) {
  __shared__ _Float16 Pl[16][32];
  const int lane = threadIdx.x;
  const int s0 = blockIdx.x * 16, g = blockIdx.y, hh = blockIdx.z;
  const int h = g * HPGc + hh;
  const int hi = lane >> 4, ln = lane & 15;

  v16h aq0 = load_a16(qh + ((size_t)s0 * Hc + h) * 64, Hc * 64, lane);
  v16h aq1 = load_a16(qh + ((size_t)s0 * Hc + h) * 64 + 32, Hc * 64, lane);

  unsigned rm[8]; unsigned lor = 0;
#pragma unroll
  for (int r = 0; r < 8; ++r) { rm[r] = selmask[(s0 + r + (hi ? 8 : 0)) * Gc + g]; lor |= rm[r]; }
  const unsigned ub = redor32(lor);

  float M[8], Ssum[8];
#pragma unroll
  for (int r = 0; r < 8; ++r) { M[r] = NEGc; Ssum[r] = 0.f; }
  v8f o0 = vzero(), o1 = vzero(), o2 = vzero(), o3 = vzero();

  for (int b = 0; b < NBc; ++b) {
    if (b * 64 > s0 + 15) break;
    if (!((ub >> b) & 1)) continue;
    for (int half = 0; half < 2; ++half) {
      int cb = b * 64 + half * 32;
      if (cb > s0 + 15) break;
      v16h kb00 = load_b16(ks + ((size_t)g * S_ + cb) * 64, 64, lane);
      v16h kb01 = load_b16(ks + ((size_t)g * S_ + cb) * 64 + 32, 64, lane);
      v16h kb10 = load_b16(ks + ((size_t)g * S_ + cb + 16) * 64, 64, lane);
      v16h kb11 = load_b16(ks + ((size_t)g * S_ + cb + 16) * 64 + 32, 64, lane);
      v8f sA = vzero(), sB = vzero();
      sA = wmma_f16(aq0, kb00, sA); sA = wmma_f16(aq1, kb01, sA);
      sB = wmma_f16(aq0, kb10, sB); sB = wmma_f16(aq1, kb11, sB);
      float nM[8], f[8];
#pragma unroll
      for (int r = 0; r < 8; ++r) {
        int s = s0 + r + (hi ? 8 : 0);
        bool sel = (rm[r] >> b) & 1;
        bool okA = sel && (cb + ln <= s);
        bool okB = sel && (cb + 16 + ln <= s);
        sA[r] = okA ? sA[r] * SCALEc : NEGc;
        sB[r] = okB ? sB[r] * SCALEc : NEGc;
        float cm = redmax16(fmaxf(sA[r], sB[r]));
        nM[r] = fmaxf(M[r], cm);
        f[r] = __expf(M[r] - nM[r]);
      }
#pragma unroll
      for (int r = 0; r < 8; ++r) {
        float pA = (sA[r] > -1e29f) ? __expf(sA[r] - nM[r]) : 0.f;
        float pB = (sB[r] > -1e29f) ? __expf(sB[r] - nM[r]) : 0.f;
        sA[r] = pA; sB[r] = pB;
        Ssum[r] = Ssum[r] * f[r] + redsum16(pA + pB);
        M[r] = nM[r];
        o0[r] *= f[r]; o1[r] *= f[r]; o2[r] *= f[r]; o3[r] *= f[r];
      }
      __syncthreads();
#pragma unroll
      for (int r = 0; r < 8; ++r) {
        Pl[r + (hi ? 8 : 0)][ln]      = (_Float16)sA[r];
        Pl[r + (hi ? 8 : 0)][16 + ln] = (_Float16)sB[r];
      }
      __syncthreads();
      v16h ap = load_a16(&Pl[0][0], 32, lane);
      v16h vb0 = load_b16(vtS + ((size_t)g * 64 +  0) * S_ + cb, S_, lane);
      v16h vb1 = load_b16(vtS + ((size_t)g * 64 + 16) * S_ + cb, S_, lane);
      v16h vb2 = load_b16(vtS + ((size_t)g * 64 + 32) * S_ + cb, S_, lane);
      v16h vb3 = load_b16(vtS + ((size_t)g * 64 + 48) * S_ + cb, S_, lane);
      o0 = wmma_f16(ap, vb0, o0);
      o1 = wmma_f16(ap, vb1, o1);
      o2 = wmma_f16(ap, vb2, o2);
      o3 = wmma_f16(ap, vb3, o3);
    }
  }
#pragma unroll
  for (int r = 0; r < 8; ++r) {
    int s = s0 + r + (hi ? 8 : 0);
    float inv = 1.0f / Ssum[r];
    float* orow = osel + ((size_t)s * Hc + h) * 64 + ln;
    orow[0] = o0[r] * inv; orow[16] = o1[r] * inv;
    orow[32] = o2[r] * inv; orow[48] = o3[r] * inv;
  }
}

// ---------------- sliding-window attention (flash over W=512 window) ----------------
__global__ void __launch_bounds__(32) k_attn_win(
    const _Float16* __restrict__ qh, const _Float16* __restrict__ kw,
    const _Float16* __restrict__ vtW, float* __restrict__ owin) {
  __shared__ _Float16 Pl[16][32];
  const int lane = threadIdx.x;
  const int s0 = blockIdx.x * 16, g = blockIdx.y, hh = blockIdx.z;
  const int h = g * HPGc + hh;
  const int hi = lane >> 4, ln = lane & 15;

  v16h aq0 = load_a16(qh + ((size_t)s0 * Hc + h) * 64, Hc * 64, lane);
  v16h aq1 = load_a16(qh + ((size_t)s0 * Hc + h) * 64 + 32, Hc * 64, lane);

  float M[8], Ssum[8];
#pragma unroll
  for (int r = 0; r < 8; ++r) { M[r] = NEGc; Ssum[r] = 0.f; }
  v8f o0 = vzero(), o1 = vzero(), o2 = vzero(), o3 = vzero();

  int lo = s0 - (Wc - 1); if (lo < 0) lo = 0;
  for (int cb = lo & ~31; cb <= s0 + 15; cb += 32) {
    v16h kb00 = load_b16(kw + ((size_t)g * S_ + cb) * 64, 64, lane);
    v16h kb01 = load_b16(kw + ((size_t)g * S_ + cb) * 64 + 32, 64, lane);
    v16h kb10 = load_b16(kw + ((size_t)g * S_ + cb + 16) * 64, 64, lane);
    v16h kb11 = load_b16(kw + ((size_t)g * S_ + cb + 16) * 64 + 32, 64, lane);
    v8f sA = vzero(), sB = vzero();
    sA = wmma_f16(aq0, kb00, sA); sA = wmma_f16(aq1, kb01, sA);
    sB = wmma_f16(aq0, kb10, sB); sB = wmma_f16(aq1, kb11, sB);
    float nM[8], f[8];
#pragma unroll
    for (int r = 0; r < 8; ++r) {
      int s = s0 + r + (hi ? 8 : 0);
      int tA = cb + ln, tB = cb + 16 + ln;
      bool okA = (tA <= s) && (tA > s - Wc);
      bool okB = (tB <= s) && (tB > s - Wc);
      sA[r] = okA ? sA[r] * SCALEc : NEGc;
      sB[r] = okB ? sB[r] * SCALEc : NEGc;
      float cm = redmax16(fmaxf(sA[r], sB[r]));
      nM[r] = fmaxf(M[r], cm);
      f[r] = __expf(M[r] - nM[r]);
    }
#pragma unroll
    for (int r = 0; r < 8; ++r) {
      float pA = (sA[r] > -1e29f) ? __expf(sA[r] - nM[r]) : 0.f;
      float pB = (sB[r] > -1e29f) ? __expf(sB[r] - nM[r]) : 0.f;
      sA[r] = pA; sB[r] = pB;
      Ssum[r] = Ssum[r] * f[r] + redsum16(pA + pB);
      M[r] = nM[r];
      o0[r] *= f[r]; o1[r] *= f[r]; o2[r] *= f[r]; o3[r] *= f[r];
    }
    __syncthreads();
#pragma unroll
    for (int r = 0; r < 8; ++r) {
      Pl[r + (hi ? 8 : 0)][ln]      = (_Float16)sA[r];
      Pl[r + (hi ? 8 : 0)][16 + ln] = (_Float16)sB[r];
    }
    __syncthreads();
    v16h ap = load_a16(&Pl[0][0], 32, lane);
    v16h vb0 = load_b16(vtW + ((size_t)g * 64 +  0) * S_ + cb, S_, lane);
    v16h vb1 = load_b16(vtW + ((size_t)g * 64 + 16) * S_ + cb, S_, lane);
    v16h vb2 = load_b16(vtW + ((size_t)g * 64 + 32) * S_ + cb, S_, lane);
    v16h vb3 = load_b16(vtW + ((size_t)g * 64 + 48) * S_ + cb, S_, lane);
    o0 = wmma_f16(ap, vb0, o0);
    o1 = wmma_f16(ap, vb1, o1);
    o2 = wmma_f16(ap, vb2, o2);
    o3 = wmma_f16(ap, vb3, o3);
  }
#pragma unroll
  for (int r = 0; r < 8; ++r) {
    int s = s0 + r + (hi ? 8 : 0);
    float inv = 1.0f / Ssum[r];
    float* orow = owin + ((size_t)s * Hc + h) * 64 + ln;
    orow[0] = o0[r] * inv; orow[16] = o1[r] * inv;
    orow[32] = o2[r] * inv; orow[48] = o3[r] * inv;
  }
}

// ---------------- gating MLP + softmax + peaked override ----------------
__global__ void k_gate(const float* __restrict__ qp, const float* __restrict__ w1,
                       const float* __restrict__ b1, const float* __restrict__ w2,
                       const float* __restrict__ b2, float* __restrict__ pgout) {
  int i = blockIdx.x * blockDim.x + threadIdx.x;  // S*G
  if (i >= S_ * Gc) return;
  const float* q = qp + (size_t)i * 64;
  float hbuf[HIDc];
  for (int j = 0; j < HIDc; ++j) {
    float z = b1[j];
    for (int d = 0; d < 64; ++d) z += q[d] * w1[j * 64 + d];
    hbuf[j] = z / (1.0f + __expf(-z));  // silu
  }
  float gv[3];
  for (int j = 0; j < 3; ++j) {
    float z = b2[j];
    for (int d = 0; d < HIDc; ++d) z += hbuf[d] * w2[j * HIDc + d];
    gv[j] = z;  // TAU == 1
  }
  float mx = fmaxf(gv[0], fmaxf(gv[1], gv[2]));
  float e0 = __expf(gv[0] - mx), e1 = __expf(gv[1] - mx), e2 = __expf(gv[2] - mx);
  float inv = 1.0f / (e0 + e1 + e2);
  float p0 = e0 * inv, p1 = e1 * inv, p2 = e2 * inv;
  int am = 0; float m1 = gv[0];
  if (gv[1] > m1) { m1 = gv[1]; am = 1; }
  if (gv[2] > m1) { m1 = gv[2]; am = 2; }
  float m2 = -3.0e38f;
  for (int j = 0; j < 3; ++j) if (j != am) m2 = fmaxf(m2, gv[j]);
  if (m1 - m2 > 50.0f) {
    p0 = (am == 0) ? 1.f : 0.f; p1 = (am == 1) ? 1.f : 0.f; p2 = (am == 2) ? 1.f : 0.f;
  }
  pgout[i * 3 + 0] = p0; pgout[i * 3 + 1] = p1; pgout[i * 3 + 2] = p2;
}

// ---------------- gated combine -> f16 ----------------
__global__ void k_combine(const float* __restrict__ ocmp, const float* __restrict__ osel,
                          const float* __restrict__ owin, const float* __restrict__ pgv,
                          _Float16* __restrict__ ocomb) {
  int i = blockIdx.x * blockDim.x + threadIdx.x;  // S*1024
  if (i >= S_ * 1024) return;
  int h = (i >> 6) & 15, s = i >> 10;
  int g = h >> 3;
  const float* p = pgv + ((size_t)s * Gc + g) * 3;
  ocomb[i] = (_Float16)(p[0] * ocmp[i] + p[1] * osel[i] + p[2] * owin[i]);
}

extern "C" void kernel_launch(void* const* d_in, const int* in_sizes, int n_in,
                              void* d_out, int out_size, void* d_ws, size_t ws_size,
                              hipStream_t stream) {
  (void)in_sizes; (void)n_in; (void)out_size; (void)ws_size;
  const float* x   = (const float*)d_in[0];
  const float* wq  = (const float*)d_in[1];
  const float* wks = (const float*)d_in[2];
  const float* wvs = (const float*)d_in[3];
  const float* wkw = (const float*)d_in[4];
  const float* wvw = (const float*)d_in[5];
  const float* wkc = (const float*)d_in[6];
  const float* wvc = (const float*)d_in[7];
  const float* wo  = (const float*)d_in[8];
  const float* f1w = (const float*)d_in[9];
  const float* f1b = (const float*)d_in[10];
  const float* f2w = (const float*)d_in[11];
  const float* f2b = (const float*)d_in[12];
  float* out = (float*)d_out;

  char* ws = (char*)d_ws;
  size_t off = 0;
  auto alloc = [&](size_t bytes) -> void* {
    void* p = ws + off; off += (bytes + 255) & ~(size_t)255; return p;
  };

  _Float16* xh    = (_Float16*)alloc((size_t)S_ * DIMc * 2);
  _Float16* wqh   = (_Float16*)alloc((size_t)1024 * 1024 * 2);
  _Float16* wkvh  = (_Float16*)alloc((size_t)768 * 1024 * 2);
  _Float16* woh   = (_Float16*)alloc((size_t)1024 * 1024 * 2);
  float*    qf    = (float*)alloc((size_t)S_ * 1024 * 4);
  float*    kvf   = (float*)alloc((size_t)S_ * 768 * 4);
  _Float16* qh    = (_Float16*)alloc((size_t)S_ * 1024 * 2);
  float*    qp    = (float*)alloc((size_t)S_ * Gc * 64 * 4);
  _Float16* ksel  = (_Float16*)alloc((size_t)Gc * S_ * 64 * 2);
  _Float16* kwin  = (_Float16*)alloc((size_t)Gc * S_ * 64 * 2);
  _Float16* kcmpr = (_Float16*)alloc((size_t)Gc * S_ * 64 * 2);
  _Float16* vselt = (_Float16*)alloc((size_t)Gc * 64 * S_ * 2);
  _Float16* vwint = (_Float16*)alloc((size_t)Gc * 64 * S_ * 2);
  _Float16* vcmpr = (_Float16*)alloc((size_t)Gc * S_ * 64 * 2);
  _Float16* kpool = (_Float16*)alloc((size_t)Gc * NCPc * 64 * 2);
  _Float16* vct   = (_Float16*)alloc((size_t)Gc * 64 * NCPc * 2);
  float*    pcmp  = (float*)alloc((size_t)S_ * Gc * HPGc * NCPc * 4);
  unsigned* selm  = (unsigned*)alloc((size_t)S_ * Gc * 4);
  float*    pgv   = (float*)alloc((size_t)S_ * Gc * 3 * 4);
  float*    ocmp  = (float*)alloc((size_t)S_ * 1024 * 4);
  float*    osel  = (float*)alloc((size_t)S_ * 1024 * 4);
  float*    owin  = (float*)alloc((size_t)S_ * 1024 * 4);
  _Float16* ocomb = (_Float16*)alloc((size_t)S_ * 1024 * 2);

  auto cvt = [&](const float* s, _Float16* d, int n) {
    k_cvt_f16<<<(n + 255) / 256, 256, 0, stream>>>(s, d, n);
  };
  cvt(x,  xh,  S_ * DIMc);
  cvt(wq, wqh, 1024 * 1024);
  cvt(wks, wkvh + 0 * 131072, 131072);
  cvt(wvs, wkvh + 1 * 131072, 131072);
  cvt(wkw, wkvh + 2 * 131072, 131072);
  cvt(wvw, wkvh + 3 * 131072, 131072);
  cvt(wkc, wkvh + 4 * 131072, 131072);
  cvt(wvc, wkvh + 5 * 131072, 131072);
  cvt(wo, woh, 1024 * 1024);

  k_gemm<<<dim3(1024 / 64, S_ / 16), 32, 0, stream>>>(xh, wqh, qf, S_, 1024, 1024);
  k_gemm<<<dim3(768 / 64, S_ / 16), 32, 0, stream>>>(xh, wkvh, kvf, S_, 768, 1024);

  k_rope_q<<<(S_ * 32 + 255) / 256, 256, 0, stream>>>(qf, qh, qp);
  k_rope_kv<<<(S_ * Gc * 32 + 255) / 256, 256, 0, stream>>>(kvf, ksel, kwin, kcmpr,
                                                            vselt, vwint, vcmpr);
  k_cmp_pool<<<(Gc * NCPc * 64 + 255) / 256, 256, 0, stream>>>(kcmpr, vcmpr, kpool, vct);

  k_attn_cmp<<<dim3(S_ / 16, Gc, HPGc), 32, 0, stream>>>(qh, kpool, vct, pcmp, ocmp);
  k_pgrp_topk<<<(S_ * Gc + 255) / 256, 256, 0, stream>>>(pcmp, selm);
  k_attn_sel<<<dim3(S_ / 16, Gc, HPGc), 32, 0, stream>>>(qh, ksel, vselt, selm, osel);
  k_attn_win<<<dim3(S_ / 16, Gc, HPGc), 32, 0, stream>>>(qh, kwin, vwint, owin);

  k_gate<<<(S_ * Gc + 255) / 256, 256, 0, stream>>>(qp, f1w, f1b, f2w, f2b, pgv);
  k_combine<<<(S_ * 1024 + 255) / 256, 256, 0, stream>>>(ocmp, osel, owin, pgv, ocomb);

  k_gemm<<<dim3(1024 / 64, S_ / 16), 32, 0, stream>>>(ocomb, woh, out, S_, 1024, 1024);
}
